// MTAttendDecoder_71914932404312
// MI455X (gfx1250) — matrix operand order repaired and checked by reading, not compiled, not run on hardware
//
#include <hip/hip_runtime.h>
#include <hip/hip_bf16.h>

// ---------------- problem constants (from reference) ----------------
#define B_      32
#define S_      64
#define C_      512
#define E_      256
#define H_      512
#define V_      32000
#define MAXLEN  32
#define START_  1
#define G3H     (3*H_)     // 1536
#define KIN0    (E_+C_)    // 768

// ---------------- CDNA5 types ----------------
typedef __attribute__((ext_vector_type(16))) __bf16        v16bf;
typedef __attribute__((ext_vector_type(8)))  float         v8f;
typedef __attribute__((ext_vector_type(4)))  unsigned int  u32x4;

__device__ __forceinline__ unsigned short f2bf(float f) {
  union { float f; unsigned int u; } x; x.f = f;
  unsigned int r = x.u + 0x7fffu + ((x.u >> 16) & 1u);   // round-nearest-even
  return (unsigned short)(r >> 16);
}

// A-matrix 16x32 bf16 tile (ISA 7.12.2): lane<16 -> M=lane, K {k0..k0+7, k0+16..k0+23}
//                                        lane>=16 -> M=lane-16, K {k0+8..15, k0+24..31}
__device__ __forceinline__ v16bf load_A(const unsigned short* base, int ld, int moff, int k0) {
  int lane = threadIdx.x & 31;
  const unsigned short* p = base + (size_t)(moff + (lane & 15)) * ld + k0 + ((lane >> 4) << 3);
  union { v16bf v; u32x4 u[2]; } r;
  r.u[0] = *(const u32x4*)(p);
  r.u[1] = *(const u32x4*)(p + 16);
  return r.v;
}

// B-matrix 32x16 bf16 tile: lane<16 -> N=lane, K {k0..k0+7, k0+16..k0+23}; lane>=16 -> N=lane-16, K {k0+8..15, k0+24..31}
// Row-major [rows x ld] weight matrix: row (n0+n) supplies B column n.
__device__ __forceinline__ v16bf load_B(const unsigned short* base, int ld, int n0, int k0) {
  int lane = threadIdx.x & 31;
  const unsigned short* p = base + (size_t)(n0 + (lane & 15)) * ld + k0 + ((lane >> 4) << 3);
  union { v16bf v; u32x4 u[2]; } r;
  r.u[0] = *(const u32x4*)(p);
  r.u[1] = *(const u32x4*)(p + 16);
  return r.v;
}

__device__ __forceinline__ v8f wmma_bf16(v16bf a, v16bf b, v8f c) {
  return __builtin_amdgcn_wmma_f32_16x16x32_bf16(false, a, false, b, (short)0, c, false, false);
}

__device__ __forceinline__ float sigmoidf_(float x) { return 1.0f / (1.0f + __expf(-x)); }

// ---------------- setup kernels ----------------
__global__ void cvt_bf16_k(const float* __restrict__ src, unsigned short* __restrict__ dst, int n) {
  int i = blockIdx.x * 256 + threadIdx.x;
  if (i < n) dst[i] = f2bf(src[i]);
}

__global__ void init_state_k(float* hf, unsigned short* hb, int* x) {
  int i = blockIdx.x * 256 + threadIdx.x;
  if (i < 4 * B_ * H_) { hf[i] = 0.0f; hb[i] = 0; }
  if (i < B_) x[i] = START_;
}

// ---------------- step kernel 1: attention + input prep ----------------
// grid=B, block=256. Writes attention weights to d_out and inp(bf16) [B x 768].
__global__ void attn_k(const float* __restrict__ ctx, const float* __restrict__ embed,
                       const float* __restrict__ Wa, const float* __restrict__ ba,
                       const int* __restrict__ x, const float* __restrict__ h1,
                       unsigned short* __restrict__ inp_b, float* __restrict__ att_out, int t) {
  int b = blockIdx.x, tid = threadIdx.x;
  __shared__ float red[256];
  __shared__ float sc[S_];
  __shared__ float hdot;

  // hidden part of the score: dot(h_top, Wa[0:H]) + ba  (scalar per batch row)
  float p = 0.0f;
  for (int k = tid; k < H_; k += 256) p += h1[b * H_ + k] * Wa[k];
  red[tid] = p; __syncthreads();
  for (int s = 128; s > 0; s >>= 1) { if (tid < s) red[tid] += red[tid + s]; __syncthreads(); }
  if (tid == 0) hdot = red[0] + ba[0];
  __syncthreads();
  float hb_ = hdot;

  // per-source scores: wave w handles s = w*8 .. w*8+7
  int wv = tid >> 5, lane = tid & 31;
  for (int si = 0; si < 8; ++si) {
    int s = wv * 8 + si;
    const float* crow = ctx + ((size_t)b * S_ + s) * C_;
    float q = 0.0f;
    for (int k = lane; k < C_; k += 32) q += crow[k] * Wa[H_ + k];
    for (int o = 16; o > 0; o >>= 1) q += __shfl_xor(q, o, 32);
    if (lane == 0) sc[s] = tanhf(hb_ + q);
  }
  __syncthreads();

  // softmax over 64 (serial by thread 0 — trivial)
  if (tid == 0) {
    float mx = sc[0];
    for (int s = 1; s < S_; ++s) mx = fmaxf(mx, sc[s]);
    float sum = 0.0f;
    for (int s = 0; s < S_; ++s) { sc[s] = __expf(sc[s] - mx); sum += sc[s]; }
    float inv = 1.0f / sum;
    for (int s = 0; s < S_; ++s) sc[s] *= inv;
  }
  __syncthreads();
  if (tid < S_) att_out[((size_t)b * MAXLEN + t) * S_ + tid] = sc[tid];

  // attend = w @ context  ->  inp[256:768]
  for (int c = tid; c < C_; c += 256) {
    float a = 0.0f;
    for (int s = 0; s < S_; ++s) a += sc[s] * ctx[((size_t)b * S_ + s) * C_ + c];
    inp_b[b * KIN0 + E_ + c] = f2bf(a);
  }
  // embedding lookup -> inp[0:256]
  int xv = x[b];
  for (int e = tid; e < E_; e += 256) inp_b[b * KIN0 + e] = f2bf(embed[(size_t)xv * E_ + e]);
}

// ---------------- step kernel 2/3: fused GRU cell via bf16 WMMA ----------------
// grid=8, block=128 (4 waves). Each wave owns one 16-wide gate-column tile j0,
// accumulates r/z/n gate GEMMs for both M-halves (batch 0-15 / 16-31), then does
// gate math in-register and writes h_new (fp32 + bf16).
__global__ void gru_k(const unsigned short* __restrict__ inp_b, int Kin,
                      const float* __restrict__ hprev_f, const unsigned short* __restrict__ hprev_b,
                      const unsigned short* __restrict__ Wih_b, const unsigned short* __restrict__ Whh_b,
                      const float* __restrict__ bih, const float* __restrict__ bhh,
                      float* __restrict__ hout_f, unsigned short* __restrict__ hout_b) {
  int wv = threadIdx.x >> 5;
  int jt = blockIdx.x * 4 + wv;          // 32 tiles over H=512
  int j0 = jt * 16;

  v8f ax[3][2]; v8f ah[3][2];
#pragma unroll
  for (int g = 0; g < 3; ++g) { ax[g][0] = (v8f){}; ax[g][1] = (v8f){}; ah[g][0] = (v8f){}; ah[g][1] = (v8f){}; }

  // gx = inp @ Wih^T   (K = Kin)
  for (int k0 = 0; k0 < Kin; k0 += 32) {
    v16bf a0 = load_A(inp_b, Kin, 0, k0);
    v16bf a1 = load_A(inp_b, Kin, 16, k0);
#pragma unroll
    for (int g = 0; g < 3; ++g) {
      v16bf bb = load_B(Wih_b, Kin, g * H_ + j0, k0);
      ax[g][0] = wmma_bf16(a0, bb, ax[g][0]);
      ax[g][1] = wmma_bf16(a1, bb, ax[g][1]);
    }
  }
  // gh = h_prev @ Whh^T  (K = 512)
  for (int k0 = 0; k0 < H_; k0 += 32) {
    v16bf a0 = load_A(hprev_b, H_, 0, k0);
    v16bf a1 = load_A(hprev_b, H_, 16, k0);
#pragma unroll
    for (int g = 0; g < 3; ++g) {
      v16bf bb = load_B(Whh_b, H_, g * H_ + j0, k0);
      ah[g][0] = wmma_bf16(a0, bb, ah[g][0]);
      ah[g][1] = wmma_bf16(a1, bb, ah[g][1]);
    }
  }

  // gate math: C/D tile layout — VGPR v, lane<16 -> M=v, lane>=16 -> M=v+8; N=lane%16
  int lane = threadIdx.x & 31;
  int n = lane & 15, half = lane >> 4;
  int j = j0 + n;
  float bir = bih[j],          bhr = bhh[j];
  float biz = bih[H_ + j],     bhz = bhh[H_ + j];
  float bin = bih[2 * H_ + j], bhn = bhh[2 * H_ + j];
#pragma unroll
  for (int v = 0; v < 8; ++v) {
#pragma unroll
    for (int ph = 0; ph < 2; ++ph) {
      int b = v + 8 * half + 16 * ph;
      float r = sigmoidf_((ax[0][ph][v] + bir) + (ah[0][ph][v] + bhr));
      float z = sigmoidf_((ax[1][ph][v] + biz) + (ah[1][ph][v] + bhz));
      float nn = tanhf((ax[2][ph][v] + bin) + r * (ah[2][ph][v] + bhn));
      float hp = hprev_f[b * H_ + j];
      float hn = (1.0f - z) * nn + z * hp;
      hout_f[b * H_ + j] = hn;
      hout_b[b * H_ + j] = f2bf(hn);
    }
  }
}

// ---------------- step kernel 4: logits GEMM [32x512]x[512x32000] via WMMA ----------------
// grid=250, block=256 (8 waves); each wave computes a 32x16 logit tile (K=512).
// A (h1 bf16, 32KB) is async-staged to LDS once per block, then read 8x from LDS
// instead of 8x from global: cuts A traffic per step from ~64MB to ~8MB.
__global__ void logits_k(const unsigned short* __restrict__ h1b, const unsigned short* __restrict__ Wob,
                         const float* __restrict__ bo, float* __restrict__ out, int t) {
  __shared__ unsigned short sA[B_ * H_];   // 32 x 512 bf16 = 32 KB
  int tid = threadIdx.x;

  // CDNA5 async global->LDS staging: 256 threads x 8 x b128 = 32 KB, coalesced.
  {
#pragma unroll
    for (int i = 0; i < 8; ++i) {
      int elt = (i * 256 + tid) * 8;                       // 8 ushorts = 16 B per issue
      unsigned int loff = (unsigned int)(size_t)(sA + elt); // generic LDS addr[31:0] == LDS offset
      unsigned long long gaddr = (unsigned long long)(size_t)(h1b + elt);
      asm volatile("global_load_async_to_lds_b128 %0, %1, off"
                   :: "v"(loff), "v"(gaddr) : "memory");
    }
    asm volatile("s_wait_asynccnt 0x0" ::: "memory");
  }
  __syncthreads();

  int wv = tid >> 5;
  int tile = blockIdx.x * 8 + wv;        // 2000 tiles over V
  int n0 = tile * 16;
  int lane = tid & 31;

  // per-lane LDS base pointers for the A tiles (funky 16-bit A layout, see load_A)
  const unsigned short* sA0 = sA + (size_t)(lane & 15) * H_ + ((lane >> 4) << 3);        // M 0..15
  const unsigned short* sA1 = sA + (size_t)(16 + (lane & 15)) * H_ + ((lane >> 4) << 3); // M 16..31

  v8f c0 = (v8f){}, c1 = (v8f){};
  for (int k0 = 0; k0 < H_; k0 += 32) {
    union { v16bf v; u32x4 u[2]; } a0, a1;
    a0.u[0] = *(const u32x4*)(sA0 + k0);
    a0.u[1] = *(const u32x4*)(sA0 + k0 + 16);
    a1.u[0] = *(const u32x4*)(sA1 + k0);
    a1.u[1] = *(const u32x4*)(sA1 + k0 + 16);
    // prefetch the next K-chunk of this wave's Wo rows (global_prefetch_b8)
    __builtin_prefetch(Wob + (size_t)(n0 + (lane & 15)) * H_ + k0 + 32, 0, 1);
    v16bf bb = load_B(Wob, H_, n0, k0);
    c0 = wmma_bf16(a0.v, bb, c0);
    c1 = wmma_bf16(a1.v, bb, c1);
  }

  int n = n0 + (lane & 15);
  int half = lane >> 4;
  float bias = bo[n];
#pragma unroll
  for (int v = 0; v < 8; ++v) {
    int b0 = v + 8 * half;               // batches 0..15
    int b1 = b0 + 16;                    // batches 16..31
    out[((size_t)b0 * MAXLEN + t) * V_ + n] = c0[v] + bias;
    out[((size_t)b1 * MAXLEN + t) * V_ + n] = c1[v] + bias;
  }
}

// ---------------- step kernel 5: row max + argmax (greedy token) ----------------
__global__ void argmax_k(const float* __restrict__ out, int t, int* __restrict__ x, float* __restrict__ maxv) {
  int b = blockIdx.x, tid = threadIdx.x;
  __shared__ float sm[512];
  __shared__ int   si[512];
  const float* base = out + ((size_t)b * MAXLEN + t) * V_;
  float m = -3.4e38f; int mi = 0;
  for (int v = tid; v < V_; v += 512) {
    float f = base[v];
    if (f > m) { m = f; mi = v; }
  }
  sm[tid] = m; si[tid] = mi; __syncthreads();
  for (int s = 256; s > 0; s >>= 1) {
    if (tid < s) {
      float fo = sm[tid + s]; int io = si[tid + s];
      if (fo > sm[tid] || (fo == sm[tid] && io < si[tid])) { sm[tid] = fo; si[tid] = io; }
    }
    __syncthreads();
  }
  if (tid == 0) { x[b] = si[0]; maxv[b] = sm[0]; }
}

// ---------------- step kernel 6: in-place log_softmax finalize ----------------
__global__ void lsm_k(float* __restrict__ out, int t, const float* __restrict__ maxv) {
  int b = blockIdx.x, tid = threadIdx.x;
  __shared__ float red[512];
  __shared__ float lsum;
  float* base = out + ((size_t)b * MAXLEN + t) * V_;
  float m = maxv[b], s = 0.0f;
  for (int v = tid; v < V_; v += 512) s += __expf(base[v] - m);
  red[tid] = s; __syncthreads();
  for (int k = 256; k > 0; k >>= 1) { if (tid < k) red[tid] += red[tid + k]; __syncthreads(); }
  if (tid == 0) lsum = __logf(red[0]);
  __syncthreads();
  float off = m + lsum;
  for (int v = tid; v < V_; v += 512) base[v] -= off;
}

// ---------------- host launch ----------------
extern "C" void kernel_launch(void* const* d_in, const int* in_sizes, int n_in,
                              void* d_out, int out_size, void* d_ws, size_t ws_size,
                              hipStream_t stream) {
  const float* ctx   = (const float*)d_in[0];
  // d_in[1] = max_len (device scalar) — reference constant 32, graph cannot branch on it
  const float* embed = (const float*)d_in[2];
  const float* Wa    = (const float*)d_in[3];
  const float* ba    = (const float*)d_in[4];
  const float* Wih0  = (const float*)d_in[5];
  const float* Whh0  = (const float*)d_in[6];
  const float* bih0  = (const float*)d_in[7];
  const float* bhh0  = (const float*)d_in[8];
  const float* Wih1  = (const float*)d_in[9];
  const float* Whh1  = (const float*)d_in[10];
  const float* bih1  = (const float*)d_in[11];
  const float* bhh1  = (const float*)d_in[12];
  const float* Wo    = (const float*)d_in[13];
  const float* bo    = (const float*)d_in[14];

  float* out_logp = (float*)d_out;
  float* out_att  = (float*)d_out + (size_t)B_ * MAXLEN * V_;

  // workspace carve-out (256B aligned)
  char* w = (char*)d_ws;
  size_t o = 0;
  auto carve = [&](size_t bytes) -> void* {
    void* p = w + o;
    o = (o + bytes + 255) & ~(size_t)255;
    return p;
  };
  unsigned short* Wo_b   = (unsigned short*)carve((size_t)V_ * H_ * 2);
  unsigned short* Wih0_b = (unsigned short*)carve((size_t)G3H * KIN0 * 2);
  unsigned short* Whh0_b = (unsigned short*)carve((size_t)G3H * H_ * 2);
  unsigned short* Wih1_b = (unsigned short*)carve((size_t)G3H * H_ * 2);
  unsigned short* Whh1_b = (unsigned short*)carve((size_t)G3H * H_ * 2);
  float*          hf     = (float*)carve((size_t)4 * B_ * H_ * 4);   // h0f[2], h1f[2]
  unsigned short* hb     = (unsigned short*)carve((size_t)4 * B_ * H_ * 2);
  unsigned short* inp_b  = (unsigned short*)carve((size_t)B_ * KIN0 * 2);
  int*            xbuf   = (int*)carve(B_ * 4);
  float*          maxv   = (float*)carve(B_ * 4);
  (void)ws_size; (void)n_in; (void)in_sizes; (void)out_size;

  // one-time (per call) weight conversion fp32 -> bf16
  {
    int n;
    n = V_ * H_;     cvt_bf16_k<<<(n + 255) / 256, 256, 0, stream>>>(Wo,   Wo_b,   n);
    n = G3H * KIN0;  cvt_bf16_k<<<(n + 255) / 256, 256, 0, stream>>>(Wih0, Wih0_b, n);
    n = G3H * H_;    cvt_bf16_k<<<(n + 255) / 256, 256, 0, stream>>>(Whh0, Whh0_b, n);
    n = G3H * H_;    cvt_bf16_k<<<(n + 255) / 256, 256, 0, stream>>>(Wih1, Wih1_b, n);
    n = G3H * H_;    cvt_bf16_k<<<(n + 255) / 256, 256, 0, stream>>>(Whh1, Whh1_b, n);
  }
  init_state_k<<<(4 * B_ * H_ + 255) / 256, 256, 0, stream>>>(hf, hb, xbuf);

  const int HB = B_ * H_;  // 16384 elems per state buffer
  for (int t = 0; t < MAXLEN; ++t) {
    int rp = t & 1, wp = 1 - rp;
    float*          h0f_r = hf + rp * HB;            float*          h0f_w = hf + wp * HB;
    float*          h1f_r = hf + 2 * HB + rp * HB;   float*          h1f_w = hf + 2 * HB + wp * HB;
    unsigned short* h0b_r = hb + rp * HB;            unsigned short* h0b_w = hb + wp * HB;
    unsigned short* h1b_r = hb + 2 * HB + rp * HB;   unsigned short* h1b_w = hb + 2 * HB + wp * HB;

    attn_k<<<B_, 256, 0, stream>>>(ctx, embed, Wa, ba, xbuf, h1f_r, inp_b, out_att, t);
    gru_k<<<8, 128, 0, stream>>>(inp_b, KIN0, h0f_r, h0b_r, Wih0_b, Whh0_b, bih0, bhh0, h0f_w, h0b_w);
    gru_k<<<8, 128, 0, stream>>>(h0b_w, H_,   h1f_r, h1b_r, Wih1_b, Whh1_b, bih1, bhh1, h1f_w, h1b_w);
    logits_k<<<V_ / 128, 256, 0, stream>>>(h1b_w, Wo_b, bo, out_logp, t);
    argmax_k<<<B_, 512, 0, stream>>>(out_logp, t, xbuf, maxv);
    lsm_k<<<B_, 512, 0, stream>>>(out_logp, t, maxv);
  }
}
